// SwapTestAttention_5978594476175
// MI455X (gfx1250) — matrix operand order repaired and checked by reading, not compiled, not run on hardware
//
#include <hip/hip_runtime.h>
#include <hip/hip_bf16.h>

#define NQ  8
#define DIM 256           // 2^NQ
#define NT  4096
#define KD  512           // 2*DIM (re|im concatenated)

typedef __attribute__((ext_vector_type(16))) _Float16 v16h;
typedef __attribute__((ext_vector_type(8)))  _Float16 v8h;
typedef __attribute__((ext_vector_type(8)))  float    v8f;

// ---------------------------------------------------------------------------
// Kernel 1: PQC statevector simulation. One block (256 threads) per token,
// blockIdx.y selects circuit: 0=Q, 1=K, 2=V. Statevector lives in LDS,
// double-buffered; each thread owns one amplitude.
// Outputs (f16, WMMA-ready):
//   Qa = [qr | qi], Qb = [qi | -qr], Kc = [kr | ki]   (each NT x KD)
//   Vm = per-qubit <Z> of V-circuit (NT x NQ, f32)
// ---------------------------------------------------------------------------
__global__ __launch_bounds__(DIM) void pqc_states_kernel(
    const float* __restrict__ x,  const float* __restrict__ pq,
    const float* __restrict__ pk, const float* __restrict__ pv,
    _Float16* __restrict__ Qa, _Float16* __restrict__ Qb,
    _Float16* __restrict__ Kc, float* __restrict__ Vm)
{
    __shared__ __align__(16) float2 buf[2][DIM];
    __shared__ float prob[DIM];

    const int i    = threadIdx.x;
    const int t    = blockIdx.x;
    const int mode = blockIdx.y;
    const float* params = (mode == 0) ? pq : (mode == 1) ? pk : pv;

    buf[0][i] = make_float2(i == 0 ? 1.0f : 0.0f, 0.0f);
    __syncthreads();
    int cur = 0;

    auto ry = [&](int q, float theta) {
        float s, c;
        __sincosf(theta * 0.5f, &s, &c);
        const int mask = 1 << (NQ - 1 - q);
        float2 a = buf[cur][i];
        float2 b = buf[cur][i ^ mask];
        float2 r;
        if ((i & mask) == 0) { r.x = c * a.x - s * b.x; r.y = c * a.y - s * b.y; }
        else                 { r.x = s * b.x + c * a.x; r.y = s * b.y + c * a.y; }
        buf[cur ^ 1][i] = r;
        __syncthreads();
        cur ^= 1;
    };
    auto rz = [&](int q, float theta) {
        const int mask = 1 << (NQ - 1 - q);
        const float ph = (i & mask) ? theta * 0.5f : -theta * 0.5f;
        float sp, cp;
        __sincosf(ph, &sp, &cp);
        float2 a = buf[cur][i];
        float2 r;
        r.x = a.x * cp - a.y * sp;
        r.y = a.x * sp + a.y * cp;
        buf[cur][i] = r;          // diagonal gate: in-place, own element only
        __syncthreads();
    };
    auto cnot = [&](int c, int tq) {
        const int cm = 1 << (NQ - 1 - c);
        const int tm = 1 << (NQ - 1 - tq);
        const int src = (i & cm) ? (i ^ tm) : i;
        float2 v = buf[cur][src];
        buf[cur ^ 1][i] = v;
        __syncthreads();
        cur ^= 1;
    };

    // data encoding
    for (int q = 0; q < NQ; ++q) ry(q, x[t * NQ + q]);
    // NL=2 layers of (RY,RZ) + ring CNOTs
    for (int l = 0; l < 2; ++l) {
        for (int q = 0; q < NQ; ++q) {
            ry(q, params[(l * NQ + q) * 2 + 0]);
            rz(q, params[(l * NQ + q) * 2 + 1]);
        }
        for (int q = 0; q < NQ; ++q) cnot(q, (q + 1) & (NQ - 1));
    }

    const float2 amp = buf[cur][i];
    const size_t base = (size_t)t * KD;
    if (mode == 0) {
        Qa[base + i]       = (_Float16)amp.x;
        Qa[base + DIM + i] = (_Float16)amp.y;
        Qb[base + i]       = (_Float16)amp.y;
        Qb[base + DIM + i] = (_Float16)(-amp.x);
    } else if (mode == 1) {
        Kc[base + i]       = (_Float16)amp.x;
        Kc[base + DIM + i] = (_Float16)amp.y;
    } else {
        prob[i] = amp.x * amp.x + amp.y * amp.y;
        __syncthreads();
        if (i < NQ) {
            float s = 0.f;
            for (int d = 0; d < DIM; ++d) {
                const float sg = ((d >> (NQ - 1 - i)) & 1) ? -1.f : 1.f;
                s += sg * prob[d];
            }
            Vm[t * NQ + i] = s;
        }
    }
}

// ---------------------------------------------------------------------------
// Kernel 2: fused swap-test attention (flash-style).
// One wave (32 lanes) per 16 query rows. For each 16-key tile:
//   Re/Im logits via 2x v_wmma_f32_16x16x32_f16 over K=512,
//   l = Re^2 + Im^2, online softmax (shfl_xor row reductions),
//   P tile -> LDS -> A-fragment, O += P·V via a third WMMA (V zero-padded).
// ---------------------------------------------------------------------------
__global__ __launch_bounds__(32) void attn_kernel(
    const _Float16* __restrict__ Qa, const _Float16* __restrict__ Qb,
    const _Float16* __restrict__ Kc, const float* __restrict__ Vm,
    float* __restrict__ out)
{
    __shared__ __align__(16) _Float16 ldsQa[16 * KD];
    __shared__ __align__(16) _Float16 ldsQb[16 * KD];
    __shared__ __align__(16) _Float16 Pld[16 * 16];

    const int lane    = threadIdx.x;
    const int half    = lane >> 4;    // 0: rows 0-7 / K-low ; 1: rows 8-15 / K-high
    const int l16     = lane & 15;
    const int rowBase = blockIdx.x * 16;

    // stage the 16 query rows (both Re/Im-combined matrices) into LDS
    {
        const uint4* sa = (const uint4*)(Qa + (size_t)rowBase * KD);
        const uint4* sb = (const uint4*)(Qb + (size_t)rowBase * KD);
        uint4* da = (uint4*)ldsQa;
        uint4* db = (uint4*)ldsQb;
        for (int it = lane; it < 16 * KD / 8; it += 32) { da[it] = sa[it]; db[it] = sb[it]; }
    }
    __syncthreads();

    float runM[8], runS[8];
    v8f o = {};
#pragma unroll
    for (int r = 0; r < 8; ++r) { runM[r] = -3.0e38f; runS[r] = 0.0f; }

    for (int j = 0; j < NT / 16; ++j) {
        v8f ca = {};
        v8f cb = {};
        const size_t krow = (size_t)(j * 16 + l16) * KD;   // B: one key row per lane
#pragma unroll
        for (int kk = 0; kk < KD / 32; ++kk) {
            const int k = kk * 32;
            // A fragment: row M=l16, K halves {0-7,16-23} (lanes<16) / {8-15,24-31}
            const int a0 = k + half * 8;
            const int a1 = k + 16 + half * 8;
            v8h qa0 = *(const v8h*)(ldsQa + l16 * KD + a0);
            v8h qa1 = *(const v8h*)(ldsQa + l16 * KD + a1);
            v8h qb0 = *(const v8h*)(ldsQb + l16 * KD + a0);
            v8h qb1 = *(const v8h*)(ldsQb + l16 * KD + a1);
            v16h aq = __builtin_shufflevector(qa0, qa1, 0,1,2,3,4,5,6,7,8,9,10,11,12,13,14,15);
            v16h ab = __builtin_shufflevector(qb0, qb1, 0,1,2,3,4,5,6,7,8,9,10,11,12,13,14,15);
            // B fragment: col N=l16, K halves {0-7,16-23} (lanes<16) / {8-15,24-31}
            const int b0 = k + half * 8;
            v8h k0 = *(const v8h*)(Kc + krow + b0);
            v8h k1 = *(const v8h*)(Kc + krow + b0 + 16);
            v16h bk = __builtin_shufflevector(k0, k1, 0,1,2,3,4,5,6,7,8,9,10,11,12,13,14,15);
            ca = __builtin_amdgcn_wmma_f32_16x16x32_f16(false, aq, false, bk, (short)0, ca, false, false);
            cb = __builtin_amdgcn_wmma_f32_16x16x32_f16(false, ab, false, bk, (short)0, cb, false, false);
        }

        // ---- online softmax over this 16x16 logit tile ----
        float p[8];
#pragma unroll
        for (int r = 0; r < 8; ++r) {
            const float l = ca[r] * ca[r] + cb[r] * cb[r];   // |overlap|^2
            float tm = l;                                     // row max over 16 cols
            tm = fmaxf(tm, __shfl_xor(tm, 1, 32));
            tm = fmaxf(tm, __shfl_xor(tm, 2, 32));
            tm = fmaxf(tm, __shfl_xor(tm, 4, 32));
            tm = fmaxf(tm, __shfl_xor(tm, 8, 32));
            const float nm   = fmaxf(runM[r], tm);
            const float corr = __expf(runM[r] - nm);
            runM[r] = nm;
            const float pe = __expf(l - nm);
            float ps = pe;                                    // row sum over 16 cols
            ps += __shfl_xor(ps, 1, 32);
            ps += __shfl_xor(ps, 2, 32);
            ps += __shfl_xor(ps, 4, 32);
            ps += __shfl_xor(ps, 8, 32);
            runS[r] = runS[r] * corr + ps;
            o[r] *= corr;                                     // rescale O accumulator
            p[r] = pe;
        }

        // ---- O += P (16x16, K-padded to 32) x V (16x8, padded to 32x16) ----
#pragma unroll
        for (int r = 0; r < 8; ++r)
            Pld[(half * 8 + r) * 16 + l16] = (_Float16)p[r];
        __syncthreads();

        v16h pa, vb;
        const int pb = half * 8;
#pragma unroll
        for (int e = 0; e < 8; ++e) {
            pa[e]     = Pld[l16 * 16 + pb + e];   // K = pb+e (valid), K>=16 -> 0
            pa[8 + e] = (_Float16)0.0f;
        }
#pragma unroll
        for (int e = 0; e < 16; ++e) {
            float v = 0.0f;
            if (e < 8 && l16 < 8) v = Vm[(j * 16 + half * 8 + e) * NQ + l16];
            vb[e] = (_Float16)v;                  // e>=8: K 16..31 are zero pad
        }
        o = __builtin_amdgcn_wmma_f32_16x16x32_f16(false, pa, false, vb, (short)0, o, false, false);
        __syncthreads();
    }

    // ---- epilogue: normalize and write (4096 x 8) ----
    if (l16 < 8) {
#pragma unroll
        for (int r = 0; r < 8; ++r) {
            const int row = rowBase + half * 8 + r;
            out[row * NQ + l16] = o[r] / runS[r];
        }
    }
}

// ---------------------------------------------------------------------------
extern "C" void kernel_launch(void* const* d_in, const int* in_sizes, int n_in,
                              void* d_out, int out_size, void* d_ws, size_t ws_size,
                              hipStream_t stream) {
    const float* x  = (const float*)d_in[0];   // (NT, NQ)
    const float* pq = (const float*)d_in[1];   // (NL, NQ, 2)
    const float* pk = (const float*)d_in[2];
    const float* pv = (const float*)d_in[3];

    // workspace layout (f16 matrices are WMMA operands)
    char* ws = (char*)d_ws;
    _Float16* Qa = (_Float16*)ws;                       // NT*KD f16 = 4 MB
    _Float16* Qb = Qa + (size_t)NT * KD;                // 4 MB
    _Float16* Kc = Qb + (size_t)NT * KD;                // 4 MB
    float*    Vm = (float*)(Kc + (size_t)NT * KD);      // NT*NQ f32 = 128 KB

    float* out = (float*)d_out;                         // (NT, NQ) f32

    dim3 g1(NT, 3, 1);
    pqc_states_kernel<<<g1, DIM, 0, stream>>>(x, pq, pk, pv, Qa, Qb, Kc, Vm);

    attn_kernel<<<NT / 16, 32, 0, stream>>>(Qa, Qb, Kc, Vm, out);
}
